// MambaBlock_8572754722875
// MI455X (gfx1250) — compile-verified, matrix operand order checked by reading, not run on hardware
//
#include <hip/hip_runtime.h>
#include <hip/hip_bf16.h>

// ---------------------------------------------------------------------------
// Mamba block forward for MI455X (gfx1250, wave32, WMMA, async global->LDS).
// ---------------------------------------------------------------------------

typedef __attribute__((ext_vector_type(16))) _Float16 v16h;
typedef __attribute__((ext_vector_type(8)))  _Float16 v8h;
typedef __attribute__((ext_vector_type(8)))  float    v8f;

// async-copy pointer types: builtin wants vector_size(16) int pointers
typedef int v4i_vs __attribute__((vector_size(16)));
typedef __attribute__((address_space(1))) v4i_vs* gas_ptr_t;
typedef __attribute__((address_space(3))) v4i_vs* las_ptr_t;

#define BATCH   4
#define SEQLEN  2048
#define DMODEL  1024
#define DINNER  2048
#define DSTATE  16
#define DTRANK  64
#define DCONV   4
#define MROWS   (BATCH * SEQLEN)        // 8192
#define NPROJ   (DTRANK + 2 * DSTATE)   // 96

#if defined(__gfx1250__) && __has_builtin(__builtin_amdgcn_global_load_async_to_lds_b128) && __has_builtin(__builtin_amdgcn_s_wait_asynccnt)
#define USE_ASYNC_LDS 1
#else
#define USE_ASYNC_LDS 0
#endif

// LDS B-slab: 64 rows x 32 halfs, padded to 40 halfs (80B) per row:
// 16B-aligned fragment loads, 20-bank row stride -> conflict-free ds_load_b128.
#define BROW_PAD 40
#define BTILE    (64 * BROW_PAD)

// ---------------------------------------------------------------------------
// 16B async copy global -> LDS (falls back to a plain uint4 copy).
// ---------------------------------------------------------------------------
__device__ __forceinline__ void copy16_g2l(const _Float16* __restrict__ g,
                                           _Float16* __restrict__ l) {
#if USE_ASYNC_LDS
    __builtin_amdgcn_global_load_async_to_lds_b128((gas_ptr_t)g, (las_ptr_t)l, 0, 0);
#else
    *(uint4*)l = *(const uint4*)g;
#endif
}

__device__ __forceinline__ void wait_async_stage() {
#if USE_ASYNC_LDS
    __builtin_amdgcn_s_wait_asynccnt(0);
#endif
    __syncthreads();
}

// ---------------------------------------------------------------------------
// WMMA fragment loader: 16x32 f16 tile from row-major memory (lda in elems).
// ISA 16-bit fragment layout: lanes 0-15 -> row=lane, K {0..7,16..23};
// lanes 16-31 -> row=lane-16, K {8..15,24..31}. Two 16B loads per lane.
// ---------------------------------------------------------------------------
__device__ __forceinline__ v16h load_frag16(const _Float16* __restrict__ base, int lda) {
    int lane = threadIdx.x & 31;
    int row  = lane & 15;
    int kc   = (lane >> 4) * 8;
    const _Float16* p = base + (size_t)row * lda + kc;
    v8h lo = *(const v8h*)(p);
    v8h hi = *(const v8h*)(p + 16);
    return __builtin_shufflevector(lo, hi, 0,1,2,3,4,5,6,7,8,9,10,11,12,13,14,15);
}

__device__ __forceinline__ v16h load_frag_lds(const _Float16* __restrict__ tile) {
    int lane = threadIdx.x & 31;
    int row  = lane & 15;
    int kc   = (lane >> 4) * 8;
    const _Float16* p = tile + row * BROW_PAD + kc;
    v8h lo = *(const v8h*)(p);
    v8h hi = *(const v8h*)(p + 16);
    return __builtin_shufflevector(lo, hi, 0,1,2,3,4,5,6,7,8,9,10,11,12,13,14,15);
}

// ---------------------------------------------------------------------------
// Blocked WMMA GEMM with LDS-staged B and async double buffering.
// Block (8 waves) -> 128x64 output tile; wave w -> rows [mb*128+w*16, +16).
// All waves share the 64x32 B slab per k-step (staged once into LDS).
// EPI: 0 = store f16   1 = store f32   2 = +bias, softplus, store f16
// ---------------------------------------------------------------------------
template <int EPI>
__global__ __launch_bounds__(256)
void gemm_wmma_lds(const _Float16* __restrict__ A, const _Float16* __restrict__ Bt,
                   void* __restrict__ Cout, int M, int N, int K,
                   const float* __restrict__ bias) {
    __shared__ _Float16 smem[2 * BTILE];

    int tid  = threadIdx.x;
    int wave = tid >> 5;
    int lane = tid & 31;
    int Nb   = N >> 6;
    int mb   = blockIdx.x / Nb;
    int nb   = blockIdx.x % Nb;
    int mt   = mb * 8 + wave;
    int nBase = nb * 64;

    const _Float16* Arow = A + (size_t)(mt * 16) * K;

    // stage one 64x32-half B slab (rows nBase..nBase+63, cols k..k+31) into buf
    auto stage = [&](int buf, int k) {
        int row   = tid >> 2;       // 0..63
        int chunk = tid & 3;        // 4 x 8-half chunks per row
        const _Float16* g = Bt + (size_t)(nBase + row) * K + k + chunk * 8;
        copy16_g2l(g, &smem[buf * BTILE + row * BROW_PAD + chunk * 8]);
    };

    stage(0, 0);
    v16h a_cur = load_frag16(Arow, K);

    v8f acc[4] = {};
    int steps = K >> 5;
    int buf = 0;
    for (int kt = 0; kt < steps; ++kt) {
        wait_async_stage();              // current slab resident + all waves synced
        v16h a_nxt = a_cur;
        if (kt + 1 < steps) {
            stage(buf ^ 1, (kt + 1) << 5);              // async fill of next slab
            a_nxt = load_frag16(Arow + ((kt + 1) << 5), K);  // overlap A load too
            __builtin_prefetch((const void*)(Arow + ((kt + 2) << 5)), 0, 1);
        }
        const _Float16* bb = &smem[buf * BTILE];
        // load all four B fragments first (one ds_load clause, partial waits),
        // then issue four back-to-back WMMAs on independent accumulators.
        v16h bfr[4];
#pragma unroll
        for (int j = 0; j < 4; ++j)
            bfr[j] = load_frag_lds(bb + j * 16 * BROW_PAD);
#pragma unroll
        for (int j = 0; j < 4; ++j)
            acc[j] = __builtin_amdgcn_wmma_f32_16x16x32_f16(
                false, a_cur, false, bfr[j], (short)0, acc[j], false, false);
        a_cur = a_nxt;
        buf ^= 1;
    }

    // C 16x16 f32 layout: VGPR i -> lanes 0-15 row=i, lanes 16-31 row=i+8.
    int rsub = (lane >> 4) * 8;
    int cc   = lane & 15;
#pragma unroll
    for (int j = 0; j < 4; ++j) {
        int col = nBase + j * 16 + cc;
#pragma unroll
        for (int i = 0; i < 8; ++i) {
            int row = mt * 16 + rsub + i;
            float v = acc[j][i];
            if constexpr (EPI == 0) {
                ((_Float16*)Cout)[(size_t)row * N + col] = (_Float16)v;
            } else if constexpr (EPI == 1) {
                ((float*)Cout)[(size_t)row * N + col] = v;
            } else {
                float z  = v + bias[col];
                float sp = (z > 20.0f) ? z : log1pf(__expf(z));
                ((_Float16*)Cout)[(size_t)row * N + col] = (_Float16)sp;
            }
        }
    }
}

// ---------------------------------------------------------------------------
// Direct (non-LDS) WMMA GEMM for narrow N (x_proj, N=96). NT n-tiles per wave.
// ---------------------------------------------------------------------------
template <int NT>
__global__ __launch_bounds__(256)
void gemm_wmma_direct(const _Float16* __restrict__ A, const _Float16* __restrict__ Bt,
                      float* __restrict__ Cout, int M, int N, int K) {
    int wave  = (blockIdx.x * (blockDim.x >> 5)) + (threadIdx.x >> 5);
    int numNg = N / (16 * NT);
    int mt    = wave / numNg;
    int ng    = wave % numNg;
    if (mt >= (M >> 4)) return;

    const _Float16* Arow = A + (size_t)(mt * 16) * K;

    v8f acc[NT] = {};
    for (int k = 0; k < K; k += 32) {
        v16h a = load_frag16(Arow + k, K);
        __builtin_prefetch((const void*)(Arow + k + 32), 0, 1);
        v16h bfr[NT];
#pragma unroll
        for (int j = 0; j < NT; ++j) {
            const _Float16* Brow = Bt + (size_t)((ng * NT + j) * 16) * K + k;
            bfr[j] = load_frag16(Brow, K);
        }
#pragma unroll
        for (int j = 0; j < NT; ++j)
            acc[j] = __builtin_amdgcn_wmma_f32_16x16x32_f16(
                false, a, false, bfr[j], (short)0, acc[j], false, false);
    }

    int lane = threadIdx.x & 31;
    int rsub = (lane >> 4) * 8;
    int cc   = lane & 15;
#pragma unroll
    for (int j = 0; j < NT; ++j) {
        int col = (ng * NT + j) * 16 + cc;
#pragma unroll
        for (int i = 0; i < 8; ++i) {
            int row = mt * 16 + rsub + i;
            Cout[(size_t)row * N + col] = acc[j][i];
        }
    }
}

// ---------------------------------------------------------------------------
// Weight convert + transpose: W[K,N] f32 -> Wt[N,K] f16
// ---------------------------------------------------------------------------
__global__ __launch_bounds__(256)
void transpose_w_kernel(const float* __restrict__ W, _Float16* __restrict__ Wt,
                        int K, int N) {
    size_t g = (size_t)blockIdx.x * blockDim.x + threadIdx.x;
    if (g >= (size_t)K * N) return;
    int    k = (int)(g % K);
    size_t n = g / K;
    Wt[n * K + k] = (_Float16)W[(size_t)k * N + n];
}

// ---------------------------------------------------------------------------
// RMSNorm: one block (8 waves) per row of [8192, 1024]
// ---------------------------------------------------------------------------
__global__ __launch_bounds__(256)
void rmsnorm_kernel(const float* __restrict__ x, const float* __restrict__ w,
                    _Float16* __restrict__ xn) {
    int row = blockIdx.x;
    const float* xr = x + (size_t)row * DMODEL;
    float ss = 0.0f;
    for (int i = threadIdx.x; i < DMODEL; i += 256) {
        float v = xr[i];
        ss += v * v;
    }
#pragma unroll
    for (int o = 16; o > 0; o >>= 1) ss += __shfl_xor(ss, o, 32);
    __shared__ float red[8];
    if ((threadIdx.x & 31) == 0) red[threadIdx.x >> 5] = ss;
    __syncthreads();
    float tot = 0.0f;
#pragma unroll
    for (int i = 0; i < 8; ++i) tot += red[i];
    float scale = rsqrtf(tot / (float)DMODEL + 1e-5f);
    _Float16* out = xn + (size_t)row * DMODEL;
    for (int i = threadIdx.x; i < DMODEL; i += 256)
        out[i] = (_Float16)(xr[i] * scale * w[i]);
}

// ---------------------------------------------------------------------------
// Causal depthwise conv1d (d_conv=4) + SiLU.
// ur16 = [8192, 4096] (u cols 0..2047, res cols 2048..4095) -> f16 [8192,2048]
// ---------------------------------------------------------------------------
__global__ __launch_bounds__(256)
void conv_silu_kernel(const _Float16* __restrict__ ur16,
                      const float* __restrict__ convw,
                      const float* __restrict__ convb,
                      _Float16* __restrict__ uconv) {
    size_t g = (size_t)blockIdx.x * blockDim.x + threadIdx.x;
    if (g >= (size_t)MROWS * DINNER) return;
    int    d = (int)(g & (DINNER - 1));
    size_t r = g >> 11;                    // row = b*SEQLEN + l
    int    l = (int)(r & (SEQLEN - 1));
    size_t b = r >> 11;
    float acc = convb[d];
#pragma unroll
    for (int k = 0; k < DCONV; ++k) {
        int ls = l + k - (DCONV - 1);
        if (ls >= 0) {
            size_t src = (b * SEQLEN + (size_t)ls) * (size_t)(2 * DINNER) + d;
            acc += convw[d * DCONV + k] * (float)ur16[src];
        }
    }
    float sil = acc / (1.0f + __expf(-acc));
    uconv[g] = (_Float16)sil;
}

// ---------------------------------------------------------------------------
// Slice dt columns (0..63) of dbc f32 [8192,96] -> dt16 f16 [8192,64]
// ---------------------------------------------------------------------------
__global__ __launch_bounds__(256)
void prep_dt_kernel(const float* __restrict__ dbc, _Float16* __restrict__ dt16) {
    size_t g = (size_t)blockIdx.x * blockDim.x + threadIdx.x;
    if (g >= (size_t)MROWS * DTRANK) return;
    size_t r = g >> 6;
    int    c = (int)(g & 63);
    dt16[g] = (_Float16)dbc[r * NPROJ + c];
}

// ---------------------------------------------------------------------------
// Selective scan: one thread per (b,d) channel; 16 states in registers.
// ---------------------------------------------------------------------------
__global__ __launch_bounds__(256)
void scan_kernel(const _Float16* __restrict__ delta16,
                 const _Float16* __restrict__ uconv16,
                 const _Float16* __restrict__ ur16,   // res in cols 2048..4095
                 const float* __restrict__ dbc,       // B cols 64..79, C 80..95
                 const float* __restrict__ A_log,
                 const float* __restrict__ Dvec,
                 _Float16* __restrict__ y16,
                 float* __restrict__ hidden_out) {
    int g = blockIdx.x * blockDim.x + threadIdx.x;   // 0..8191
    int d = g & (DINNER - 1);
    int b = g >> 11;

    float Arow[DSTATE];
#pragma unroll
    for (int s = 0; s < DSTATE; ++s) Arow[s] = -__expf(A_log[(size_t)d * DSTATE + s]);
    float Dd = Dvec[d];

    float h[DSTATE];
#pragma unroll
    for (int s = 0; s < DSTATE; ++s) h[s] = 0.0f;

    for (int t = 0; t < SEQLEN; ++t) {
        size_t r  = (size_t)b * SEQLEN + t;
        float  dl = (float)delta16[r * DINNER + d];
        float  u  = (float)uconv16[r * DINNER + d];
        const float* BC = dbc + r * NPROJ + DTRANK;  // wave-uniform address
        float du = dl * u;
        float y  = 0.0f;
#pragma unroll
        for (int s = 0; s < DSTATE; ++s) {
            float dA = __expf(dl * Arow[s]);
            h[s] = dA * h[s] + du * BC[s];
            y   += h[s] * BC[DSTATE + s];
        }
        y += u * Dd;
        float res = (float)ur16[r * (size_t)(2 * DINNER) + DINNER + d];
        float gate = res / (1.0f + __expf(-res));
        y16[r * DINNER + d] = (_Float16)(y * gate);
    }

    float* ho = hidden_out + (size_t)g * DSTATE;     // hidden[b, d, s]
#pragma unroll
    for (int s = 0; s < DSTATE; ++s) ho[s] = h[s];
}

// ---------------------------------------------------------------------------
extern "C" void kernel_launch(void* const* d_in, const int* in_sizes, int n_in,
                              void* d_out, int out_size, void* d_ws, size_t ws_size,
                              hipStream_t stream) {
    (void)in_sizes; (void)n_in; (void)out_size; (void)ws_size;
    const float* x         = (const float*)d_in[0];
    const float* norm_w    = (const float*)d_in[1];
    const float* in_proj_w = (const float*)d_in[2];   // [1024, 4096]
    const float* conv_w    = (const float*)d_in[3];   // [2048, 4]
    const float* conv_b    = (const float*)d_in[4];
    const float* x_proj_w  = (const float*)d_in[5];   // [2048, 96]
    const float* dt_proj_w = (const float*)d_in[6];   // [64, 2048]
    const float* dt_proj_b = (const float*)d_in[7];
    const float* A_log     = (const float*)d_in[8];   // [2048, 16]
    const float* Dvec      = (const float*)d_in[9];
    const float* out_proj_w= (const float*)d_in[10];  // [2048, 1024]

    char*  ws  = (char*)d_ws;
    size_t off = 0;
    auto alloc = [&](size_t bytes) -> void* {
        off = (off + 255) & ~(size_t)255;
        void* p = ws + off;
        off += bytes;
        return p;
    };

    _Float16* xn16     = (_Float16*)alloc((size_t)MROWS * DMODEL * 2);
    _Float16* WinT     = (_Float16*)alloc((size_t)(2 * DINNER) * DMODEL * 2); // [4096,1024]
    _Float16* XprojT   = (_Float16*)alloc((size_t)NPROJ * DINNER * 2);        // [96,2048]
    _Float16* DtprojT  = (_Float16*)alloc((size_t)DINNER * DTRANK * 2);       // [2048,64]
    _Float16* OutprojT = (_Float16*)alloc((size_t)DMODEL * DINNER * 2);       // [1024,2048]
    _Float16* ur16     = (_Float16*)alloc((size_t)MROWS * (2 * DINNER) * 2);  // [8192,4096]
    _Float16* uconv16  = (_Float16*)alloc((size_t)MROWS * DINNER * 2);
    float*    dbc      = (float*)   alloc((size_t)MROWS * NPROJ * 4);
    _Float16* dt16     = (_Float16*)alloc((size_t)MROWS * DTRANK * 2);
    _Float16* delta16  = (_Float16*)alloc((size_t)MROWS * DINNER * 2);
    _Float16* y16      = (_Float16*)alloc((size_t)MROWS * DINNER * 2);

    float* out_f32    = (float*)d_out;                       // [8192,1024]
    float* hidden_out = out_f32 + (size_t)MROWS * DMODEL;    // [4,2048,16]

    // 1) weight transposes (f32 -> f16, [K,N] -> [N,K])
    {
        size_t n;
        n = (size_t)DMODEL * (2 * DINNER);
        transpose_w_kernel<<<(unsigned)((n + 255) / 256), 256, 0, stream>>>(
            in_proj_w, WinT, DMODEL, 2 * DINNER);
        n = (size_t)DINNER * NPROJ;
        transpose_w_kernel<<<(unsigned)((n + 255) / 256), 256, 0, stream>>>(
            x_proj_w, XprojT, DINNER, NPROJ);
        n = (size_t)DTRANK * DINNER;
        transpose_w_kernel<<<(unsigned)((n + 255) / 256), 256, 0, stream>>>(
            dt_proj_w, DtprojT, DTRANK, DINNER);
        n = (size_t)DINNER * DMODEL;
        transpose_w_kernel<<<(unsigned)((n + 255) / 256), 256, 0, stream>>>(
            out_proj_w, OutprojT, DINNER, DMODEL);
    }

    // 2) RMSNorm
    rmsnorm_kernel<<<MROWS, 256, 0, stream>>>(x, norm_w, xn16);

    // 3) in_proj: [8192,1024] x [1024,4096] -> ur16 f16   (128x64 blocks)
    gemm_wmma_lds<0><<<(MROWS / 128) * ((2 * DINNER) / 64), 256, 0, stream>>>(
        xn16, WinT, (void*)ur16, MROWS, 2 * DINNER, DMODEL, nullptr);

    // 4) conv + SiLU
    {
        size_t n = (size_t)MROWS * DINNER;
        conv_silu_kernel<<<(unsigned)((n + 255) / 256), 256, 0, stream>>>(
            ur16, conv_w, conv_b, uconv16);
    }

    // 5) x_proj: [8192,2048] x [2048,96] -> dbc f32 (narrow N, direct kernel)
    {
        int waves = (MROWS / 16) * (NPROJ / 32);
        gemm_wmma_direct<2><<<waves / 8, 256, 0, stream>>>(
            uconv16, XprojT, dbc, MROWS, NPROJ, DINNER);
    }

    // 6) dt slice -> f16
    {
        size_t n = (size_t)MROWS * DTRANK;
        prep_dt_kernel<<<(unsigned)((n + 255) / 256), 256, 0, stream>>>(dbc, dt16);
    }

    // 7) dt_proj + bias + softplus: [8192,64] x [64,2048] -> delta16
    gemm_wmma_lds<2><<<(MROWS / 128) * (DINNER / 64), 256, 0, stream>>>(
        dt16, DtprojT, (void*)delta16, MROWS, DINNER, DTRANK, dt_proj_b);

    // 8) selective scan (+skip, +gate) -> y16, hidden -> d_out tail
    scan_kernel<<<(MROWS + 255) / 256, 256, 0, stream>>>(
        delta16, uconv16, ur16, dbc, A_log, Dvec, y16, hidden_out);

    // 9) out_proj: [8192,2048] x [2048,1024] -> d_out f32
    gemm_wmma_lds<1><<<(MROWS / 128) * (DMODEL / 64), 256, 0, stream>>>(
        y16, OutprojT, (void*)out_f32, MROWS, DMODEL, DINNER, nullptr);
}
